// Head_42150809043393
// MI455X (gfx1250) — matrix operand order repaired and checked by reading, not compiled, not run on hardware
//
#include <hip/hip_runtime.h>

// ---------------------------------------------------------------------------
// Causal attention head for MI455X (gfx1250, wave32, WMMA).
//   x:[4,4096,1024] f32 ; Wk/Wq/Wv:[1024,64] f32 ; bk/bq/bv:[64] f32
//   out:[4,4096,64] f32
// All matmuls run on v_wmma_f32_16x16x32_bf16 (fp32 accumulate).
// Compute-bound problem (~15 GFLOP vs ~75 MB of HBM traffic -> 3 us at
// 23.3 TB/s); K/V/W are L2/WGP$-resident, so both GEMM kernels are
// WMMA-issue/latency bound -> software double-buffering everywhere.
// ---------------------------------------------------------------------------

typedef __attribute__((ext_vector_type(16))) __bf16        v16bf;
typedef __attribute__((ext_vector_type(8)))  float         v8f;
typedef __attribute__((ext_vector_type(8)))  unsigned int  v8u;
typedef __attribute__((ext_vector_type(4)))  unsigned int  v4u;
typedef __attribute__((ext_vector_type(4)))  float         v4f;

#define EMBED 1024
#define HEAD  64
#define NB    4
#define SEQ   4096

__device__ __forceinline__ v16bf mk16(v4u lo, v4u hi) {
  v8u w = __builtin_shufflevector(lo, hi, 0, 1, 2, 3, 4, 5, 6, 7);
  return __builtin_bit_cast(v16bf, w);
}
// Native f32 -> bf16 (lowers to v_cvt_pk_bf16_f32 on gfx1250).
__device__ __forceinline__ v16bf cvtA(v4f f0, v4f f1, v4f f2, v4f f3) {
  v16bf a;
  a[0] = (__bf16)f0[0];  a[1] = (__bf16)f0[1];  a[2]  = (__bf16)f0[2];  a[3]  = (__bf16)f0[3];
  a[4] = (__bf16)f1[0];  a[5] = (__bf16)f1[1];  a[6]  = (__bf16)f1[2];  a[7]  = (__bf16)f1[3];
  a[8] = (__bf16)f2[0];  a[9] = (__bf16)f2[1];  a[10] = (__bf16)f2[2];  a[11] = (__bf16)f2[3];
  a[12] = (__bf16)f3[0]; a[13] = (__bf16)f3[1]; a[14] = (__bf16)f3[2];  a[15] = (__bf16)f3[3];
  return a;
}

// ---------------------------------------------------------------------------
// Kernel 0: W[1024,64] f32 -> Wt[mat][64][1024] bf16 (contraction-contiguous
// WMMA B operands).
// ---------------------------------------------------------------------------
__global__ void prep_w(const float* __restrict__ Wq, const float* __restrict__ Wk,
                       const float* __restrict__ Wv, __bf16* __restrict__ Wt) {
  int i = blockIdx.x * blockDim.x + threadIdx.x;       // [0, 3*64*1024)
  if (i >= 3 * HEAD * EMBED) return;
  int mat = i / (HEAD * EMBED);
  int rem = i - mat * (HEAD * EMBED);
  int h = rem / EMBED;
  int c = rem - h * EMBED;
  const float* W = (mat == 0) ? Wq : ((mat == 1) ? Wk : Wv);
  Wt[i] = (__bf16)W[c * HEAD + h];                     // Wt[mat][h][c]
}

// ---------------------------------------------------------------------------
// Kernel 1: QKV projection.  One wave = one 16-row tile of one matrix, full
// N=64 (4 accumulator tiles), K=1024 in 32 chunks of 32.  Both the weight
// (B) tiles and the x (A) chunk are double-buffered one K-chunk ahead so the
// four WMMAs of chunk kk overlap the loads of chunk kk+1.  Q,K stored bf16
// row-major [16384][64]; V stored transposed Vt[b][64][4096] bf16.
// ---------------------------------------------------------------------------
__global__ void __launch_bounds__(128) qkv_proj(
    const float* __restrict__ x, const __bf16* __restrict__ Wt,
    const float* __restrict__ bq, const float* __restrict__ bk,
    const float* __restrict__ bv, __bf16* __restrict__ Qb,
    __bf16* __restrict__ Kb, __bf16* __restrict__ Vt) {
  const int lane = threadIdx.x & 31;
  const int wid  = threadIdx.x >> 5;
  const int gw   = blockIdx.x * 4 + wid;   // [0, 3072)
  const int mat  = gw >> 10;               // 0=q, 1=k, 2=v
  const int mt   = gw & 1023;              // 16-row tile index

  const int l15  = lane & 15;
  const int hi   = lane >> 4;              // half-wave select
  const int koff = hi * 8;                 // unused; A and B both use kb2
  const int kb2  = hi * 16;                // A/B K offset (contiguous k = 16*hi + s)

  const v8f z8 = {0.f, 0.f, 0.f, 0.f, 0.f, 0.f, 0.f, 0.f};
  v8f acc[4] = {z8, z8, z8, z8};

  const float* xrow = x + (size_t)(mt * 16 + l15) * EMBED;
  const __bf16* wbase = Wt + (size_t)mat * HEAD * EMBED;

  auto loadB = [&](int kk, v16bf* b) {
#pragma unroll
    for (int n = 0; n < 4; ++n) {
      const __bf16* wp = wbase + (size_t)(n * 16 + l15) * EMBED + kk * 32 + kb2;
      b[n] = mk16(*(const v4u*)wp, *(const v4u*)(wp + 8));
    }
  };
  auto loadX = [&](int kk, v4f* f) {
    const float* p0 = xrow + kk * 32 + kb2;
    const float* p1 = xrow + kk * 32 + 8 + kb2;
    f[0] = *(const v4f*)(p0); f[1] = *(const v4f*)(p0 + 4);
    f[2] = *(const v4f*)(p1); f[3] = *(const v4f*)(p1 + 4);
  };

  v16bf bc[4], bn[4];
  v4f   fc[4], fn[4];
  loadB(0, bc);
  loadX(0, fc);

#pragma unroll 2
  for (int kk = 0; kk < EMBED / 32; ++kk) {
    const int kkn = (kk < EMBED / 32 - 1) ? kk + 1 : kk;  // branchless clamp
    loadB(kkn, bn);
    loadX(kkn, fn);
    v16bf a = cvtA(fc[0], fc[1], fc[2], fc[3]);
#pragma unroll
    for (int n = 0; n < 4; ++n)
      acc[n] = __builtin_amdgcn_wmma_f32_16x16x32_bf16(
          false, a, false, bc[n], (short)0, acc[n], false, false);
    // rotate double buffers (copy-propagated under unroll-2)
#pragma unroll
    for (int n = 0; n < 4; ++n) bc[n] = bn[n];
#pragma unroll
    for (int n = 0; n < 4; ++n) fc[n] = fn[n];
  }

  const float* bias = (mat == 0) ? bq : ((mat == 1) ? bk : bv);
#pragma unroll
  for (int n = 0; n < 4; ++n) {
    const int col = n * 16 + l15;
    const float bval = bias[col];
#pragma unroll
    for (int r = 0; r < 8; ++r) {
      const int row = mt * 16 + r + hi * 8;          // global row [0,16384)
      const __bf16 h16 = (__bf16)(acc[n][r] + bval);
      if (mat == 0)      Qb[(size_t)row * HEAD + col] = h16;
      else if (mat == 1) Kb[(size_t)row * HEAD + col] = h16;
      else {
        const int bb = row >> 12, t = row & (SEQ - 1);
        Vt[((size_t)(bb * HEAD + col)) * SEQ + t] = h16;  // transposed
      }
    }
  }
}

// ---------------------------------------------------------------------------
// Kernel 2: flash attention.  One wave = one (batch, 16-query tile); each
// block holds the 4 batches of one qt (uniform trip count), qt = 255-blockIdx
// so longest tiles schedule first.  Per 32-key tile: 4 WMMAs for S = Q K^T
// (K tile software-double-buffered), online softmax with per-tile max
// reduction only (row-sum reduced once in epilogue), P re-layout via LDS,
// 4 WMMAs for O = alpha*O + P V (V loaded at top of iteration, consumed
// after softmax -> latency naturally covered).
// ---------------------------------------------------------------------------
__global__ void __launch_bounds__(128) attn(
    const __bf16* __restrict__ Qb, const __bf16* __restrict__ Kb,
    const __bf16* __restrict__ Vt, float* __restrict__ out) {
  __shared__ __align__(16) __bf16 lds[4][16 * 32];

  const int lane = threadIdx.x & 31;
  const int wid  = threadIdx.x >> 5;
  const int bb   = wid;                    // batch (4 waves = 4 batches)
  const int qt   = 255 - blockIdx.x;       // longest tiles first

  const int l15  = lane & 15;
  const int hi   = lane >> 4;
  const int koff = hi * 8;
  const int kb2  = hi * 16;

  // Q A-operands for both K=32 head-dim chunks: loop invariant.
  const __bf16* qrow = Qb + ((size_t)(bb * SEQ + qt * 16 + l15)) * HEAD;
  const v16bf qa0 = mk16(*(const v4u*)(qrow + kb2),
                         *(const v4u*)(qrow + 8 + kb2));
  const v16bf qa1 = mk16(*(const v4u*)(qrow + 32 + kb2),
                         *(const v4u*)(qrow + 40 + kb2));

  const v8f z8 = {0.f, 0.f, 0.f, 0.f, 0.f, 0.f, 0.f, 0.f};
  v8f o[4] = {z8, z8, z8, z8};
  float m[8], lp[8];                       // running max, per-lane partial sum
#pragma unroll
  for (int r = 0; r < 8; ++r) { m[r] = -1e30f; lp[r] = 0.f; }

  const __bf16* kbb = Kb + (size_t)(bb * SEQ) * HEAD;
  const __bf16* vbb = Vt + (size_t)(bb * HEAD) * SEQ;
  __bf16* myl = lds[wid];
  const int ktmax = (qt * 16 + 15) >> 5;

  auto loadK = [&](int kt, v16bf* kb) {
#pragma unroll
    for (int st = 0; st < 2; ++st) {
      const __bf16* krow = kbb + (size_t)(kt * 32 + st * 16 + l15) * HEAD;
      kb[st * 2 + 0] = mk16(*(const v4u*)(krow + kb2),
                            *(const v4u*)(krow + kb2 + 8));
      kb[st * 2 + 1] = mk16(*(const v4u*)(krow + 32 + kb2),
                            *(const v4u*)(krow + 40 + kb2));
    }
  };

  v16bf kc[4], kn[4];
  loadK(0, kc);

#pragma unroll 2
  for (int kt = 0; kt <= ktmax; ++kt) {
    // Issue V loads for this tile now; consumed only after softmax.
    v16bf vc[4];
#pragma unroll
    for (int n = 0; n < 4; ++n) {
      const __bf16* vrow = vbb + (size_t)(n * 16 + l15) * SEQ + kt * 32 + kb2;
      vc[n] = mk16(*(const v4u*)vrow, *(const v4u*)(vrow + 8));
    }
    // Branchless prefetch of next K tile (clamped -> always in bounds).
    const int ktn = (kt < ktmax) ? kt + 1 : ktmax;
    loadK(ktn, kn);

    // ---- S = (Q K^T) * scale, causal mask ------------------------------
    v8f s[2];
#pragma unroll
    for (int st = 0; st < 2; ++st) {
      v8f a = z8;
      a = __builtin_amdgcn_wmma_f32_16x16x32_bf16(
          false, qa0, false, kc[st * 2 + 0], (short)0, a, false, false);
      a = __builtin_amdgcn_wmma_f32_16x16x32_bf16(
          false, qa1, false, kc[st * 2 + 1], (short)0, a, false, false);
      s[st] = a;
    }
    const int rowbase = qt * 16 + hi * 8;
#pragma unroll
    for (int st = 0; st < 2; ++st) {
      const int key = kt * 32 + st * 16 + l15;
#pragma unroll
      for (int r = 0; r < 8; ++r)
        s[st][r] = (key <= rowbase + r) ? s[st][r] * 0.125f : -1e30f;
    }

    // ---- online softmax (max reduction per tile; sum stays per-lane) ---
    float alpha[8];
#pragma unroll
    for (int r = 0; r < 8; ++r) {
      float tm = fmaxf(s[0][r], s[1][r]);
      tm = fmaxf(tm, __shfl_xor(tm, 1, 32));
      tm = fmaxf(tm, __shfl_xor(tm, 2, 32));
      tm = fmaxf(tm, __shfl_xor(tm, 4, 32));
      tm = fmaxf(tm, __shfl_xor(tm, 8, 32));
      const float mn = fmaxf(m[r], tm);
      alpha[r] = __expf(m[r] - mn);
      m[r] = mn;
      const float p0 = __expf(s[0][r] - mn);
      const float p1 = __expf(s[1][r] - mn);
      s[0][r] = p0; s[1][r] = p1;
      lp[r] = lp[r] * alpha[r] + (p0 + p1);
    }

    // ---- P (C/D layout) -> LDS -> A layout -----------------------------
#pragma unroll
    for (int st = 0; st < 2; ++st) {
      const int col = st * 16 + l15;
#pragma unroll
      for (int r = 0; r < 8; ++r)
        myl[(r + hi * 8) * 32 + col] = (__bf16)s[st][r];
    }
    asm volatile("s_wait_dscnt 0x0" ::: "memory");  // DS in-order per wave
    const v16bf pa = mk16(*(const v4u*)&myl[l15 * 32 + kb2],
                          *(const v4u*)&myl[l15 * 32 + 8 + kb2]);

    // ---- O = alpha*O + P V --------------------------------------------
#pragma unroll
    for (int n = 0; n < 4; ++n)
#pragma unroll
      for (int r = 0; r < 8; ++r) o[n][r] *= alpha[r];
#pragma unroll
    for (int n = 0; n < 4; ++n)
      o[n] = __builtin_amdgcn_wmma_f32_16x16x32_bf16(
          false, pa, false, vc[n], (short)0, o[n], false, false);

    // rotate K double-buffer (copy-propagated under unroll-2)
#pragma unroll
    for (int i = 0; i < 4; ++i) kc[i] = kn[i];
  }

  // ---- epilogue: reduce row sums once, O * (1/l), fp32 store -----------
  float inv[8];
#pragma unroll
  for (int r = 0; r < 8; ++r) {
    float rs = lp[r];
    rs += __shfl_xor(rs, 1, 32);
    rs += __shfl_xor(rs, 2, 32);
    rs += __shfl_xor(rs, 4, 32);
    rs += __shfl_xor(rs, 8, 32);
    inv[r] = __builtin_amdgcn_rcpf(rs);
  }
#pragma unroll
  for (int n = 0; n < 4; ++n) {
    const int col = n * 16 + l15;
#pragma unroll
    for (int r = 0; r < 8; ++r) {
      const int row = bb * SEQ + qt * 16 + r + hi * 8;
      out[(size_t)row * HEAD + col] = o[n][r] * inv[r];
    }
  }
}

// ---------------------------------------------------------------------------
// Host-side launcher.  Workspace layout (bf16):
//   Qb  [16384][64]      @ 0        (2 MB)
//   Kb  [16384][64]      @ 2 MB
//   Vt  [4][64][4096]    @ 4 MB
//   Wt  [3][64][1024]    @ 6 MB     (384 KB)
// ---------------------------------------------------------------------------
extern "C" void kernel_launch(void* const* d_in, const int* in_sizes, int n_in,
                              void* d_out, int out_size, void* d_ws, size_t ws_size,
                              hipStream_t stream) {
  const float* x  = (const float*)d_in[0];
  const float* Wk = (const float*)d_in[1];
  const float* bk = (const float*)d_in[2];
  const float* Wq = (const float*)d_in[3];
  const float* bq = (const float*)d_in[4];
  const float* Wv = (const float*)d_in[5];
  const float* bv = (const float*)d_in[6];

  char* ws = (char*)d_ws;
  __bf16* Qb = (__bf16*)(ws);
  __bf16* Kb = (__bf16*)(ws + (size_t)2 * 1024 * 1024);
  __bf16* Vt = (__bf16*)(ws + (size_t)4 * 1024 * 1024);
  __bf16* Wt = (__bf16*)(ws + (size_t)6 * 1024 * 1024);

  prep_w<<<768, 256, 0, stream>>>(Wq, Wk, Wv, Wt);
  qkv_proj<<<768, 128, 0, stream>>>(x, Wt, bq, bk, bv, Qb, Kb, Vt);
  attn<<<256, 128, 0, stream>>>(Qb, Kb, Vt, (float*)d_out);
}
